// DecoderV13_47064251630132
// MI455X (gfx1250) — compile-verified, matrix operand order checked by reading, not run on hardware
//
#include <hip/hip_runtime.h>
#include <hip/hip_bf16.h>

typedef __attribute__((ext_vector_type(16))) _Float16 v16h;
typedef __attribute__((ext_vector_type(8)))  _Float16 v8h;
typedef __attribute__((ext_vector_type(8)))  float    v8f;
typedef unsigned int u32x4 __attribute__((ext_vector_type(4)));
typedef int          i32x8 __attribute__((ext_vector_type(8)));
typedef int          i32x4 __attribute__((ext_vector_type(4)));

#define GRID_H 256
#define GRID_W 512
#define HW_ (GRID_H * GRID_W)
#define CHUNK 20000

// ---------------------------------------------------------------------------
// Tiled transpose: grid (C=256, HW) f32 -> grid_t (HW, C). One-time 256MB of
// streaming traffic (~11us at 23.3 TB/s) converts the per-query 4KB gather
// from 1024 scattered dwords into 4 contiguous 1KB rows.
// ---------------------------------------------------------------------------
__global__ __launch_bounds__(256) void transpose_kernel(
    const float* __restrict__ src, float* __restrict__ dst)
{
    __shared__ float tile[32][33];
    int tx = threadIdx.x & 31;
    int ty = threadIdx.x >> 5;          // 0..7
    size_t x0 = (size_t)blockIdx.x * 32; // HW index
    size_t c0 = (size_t)blockIdx.y * 32; // channel index
#pragma unroll
    for (int i = 0; i < 32; i += 8)
        tile[ty + i][tx] = src[(c0 + ty + i) * (size_t)HW_ + x0 + tx];
    __syncthreads();
#pragma unroll
    for (int i = 0; i < 32; i += 8)
        dst[(x0 + ty + i) * 256 + c0 + tx] = tile[tx][ty + i];
}

// ---------------------------------------------------------------------------
// Pack f32 weight matrix (Kreal x Nreal, row-major) into WMMA B-fragment tiles.
// Per CDNA5 ISA 16-bit B layout (wave32):
//   lane l<16 : col = nt*16+l,      halves 0..7 -> K 0..7,  8..15 -> K 16..23
//   lane l>=16: col = nt*16+(l-16), halves 0..7 -> K 8..15, 8..15 -> K 24..31
// Stored lane-major: dst[tile*512 + lane*16 + h], tile = nt*KB + kb (1KB/tile).
// ---------------------------------------------------------------------------
__global__ void pack_kernel(const float* __restrict__ src, _Float16* __restrict__ dst,
                            int Kreal, int Nreal, int KB, int NT)
{
    int t    = blockIdx.x * 256 + threadIdx.x;
    int lane = t & 31;
    int tile = t >> 5;
    if (tile >= KB * NT) return;
    int kb = tile % KB;
    int nt = tile / KB;
    int n    = nt * 16 + (lane & 15);
    int kofs = (lane < 16) ? 0 : 8;
    _Float16* d = dst + (size_t)tile * 512 + (size_t)lane * 16;
#pragma unroll
    for (int h = 0; h < 16; ++h) {
        int k = kb * 32 + ((h < 8) ? (kofs + h) : (16 + kofs + (h - 8)));
        float v = (k < Kreal && n < Nreal) ? src[(size_t)k * Nreal + n] : 0.0f;
        d[h] = (_Float16)v;
    }
}

// ---------------------------------------------------------------------------
// Per-query prep: fourier features (231 -> padded 256, f16) and 256-channel
// bilinear gather (from transposed grid) into ctx cols [0,256). One wave/query;
// lane handles 8 contiguous channels -> two b128 loads per corner, coalesced.
// ---------------------------------------------------------------------------
__global__ __launch_bounds__(256) void prep_kernel(
    const float* __restrict__ gt,    // (HW, 256) transposed grid
    const float* __restrict__ sgrad, // (2, H, W)
    const float* __restrict__ qpos, const float* __restrict__ quinf,
    const float* __restrict__ qsdf, const float* __restrict__ qnorm,
    const float* __restrict__ qflow,
    _Float16* __restrict__ ff, _Float16* __restrict__ ctx,
    int base, int rows, int rows16)
{
    int wave = threadIdx.x >> 5;
    int lane = threadIdx.x & 31;
    int q = blockIdx.x * 8 + wave;
    if (q >= rows16) return;
    if (q >= rows) {  // zero-pad tail rows so GEMM tiles are clean
        for (int j = lane; j < 256; j += 32) ff[(size_t)q * 256 + j] = (_Float16)0.0f;
        for (int c = lane; c < 256; c += 32) ctx[(size_t)q * 768 + c] = (_Float16)0.0f;
        return;
    }
    int gq = base + q;
    float px = qpos[2 * gq + 0], py = qpos[2 * gq + 1];
    float xn = 2.0f * (px - (-2.0f)) / 6.0f - 1.0f;
    float yn = 2.0f * (py - (-1.5f)) / 3.0f - 1.0f;
    xn = fminf(1.0f, fmaxf(-1.0f, xn));
    yn = fminf(1.0f, fmaxf(-1.0f, yn));
    float ix = (xn + 1.0f) * 0.5f * (float)(GRID_W - 1);
    float iy = (yn + 1.0f) * 0.5f * (float)(GRID_H - 1);
    float x0f = floorf(ix), y0f = floorf(iy);
    float wx = ix - x0f, wy = iy - y0f;
    int x0 = (int)fminf((float)(GRID_W - 1), fmaxf(0.0f, x0f));
    int x1 = (int)fminf((float)(GRID_W - 1), fmaxf(0.0f, x0f + 1.0f));
    int y0 = (int)fminf((float)(GRID_H - 1), fmaxf(0.0f, y0f));
    int y1 = (int)fminf((float)(GRID_H - 1), fmaxf(0.0f, y0f + 1.0f));
    float w00 = (1.0f - wx) * (1.0f - wy), w01 = wx * (1.0f - wy);
    float w10 = (1.0f - wx) * wy,          w11 = wx * wy;
    int i00 = y0 * GRID_W + x0, i01 = y0 * GRID_W + x1;
    int i10 = y1 * GRID_W + x0, i11 = y1 * GRID_W + x1;

    float id[11];
    id[0] = px; id[1] = py;
    id[2] = quinf[2 * gq]; id[3] = quinf[2 * gq + 1];
    id[4] = qsdf[gq];
    id[5] = w00 * sgrad[i00] + w01 * sgrad[i01] + w10 * sgrad[i10] + w11 * sgrad[i11];
    id[6] = w00 * sgrad[HW_ + i00] + w01 * sgrad[HW_ + i01] +
            w10 * sgrad[HW_ + i10] + w11 * sgrad[HW_ + i11];
    id[7] = qnorm[2 * gq]; id[8] = qnorm[2 * gq + 1];
    id[9] = qflow[2 * gq]; id[10] = qflow[2 * gq + 1];

    const float PI_ = 3.14159265358979323846f;
    for (int j = lane; j < 256; j += 32) {
        float v;
        if (j < 11) {
            v = id[j];
        } else if (j < 121) {
            int k = j - 11; int d = k / 10; int f = k % 10;
            v = sinf(id[d] * PI_ * (float)(1 << f));
        } else if (j < 231) {
            int k = j - 121; int d = k / 10; int f = k % 10;
            v = cosf(id[d] * PI_ * (float)(1 << f));
        } else {
            v = 0.0f;
        }
        ff[(size_t)q * 256 + j] = (_Float16)v;
    }

    // Coalesced 256-channel bilinear gather from transposed grid
    const float* g00 = gt + (size_t)i00 * 256 + lane * 8;
    const float* g01 = gt + (size_t)i01 * 256 + lane * 8;
    const float* g10 = gt + (size_t)i10 * 256 + lane * 8;
    const float* g11 = gt + (size_t)i11 * 256 + lane * 8;
    v8h hv;
#pragma unroll
    for (int u = 0; u < 8; ++u) {
        float v = w00 * g00[u] + w01 * g01[u] + w10 * g10[u] + w11 * g11[u];
        hv[u] = (_Float16)v;
    }
    *(v8h*)(ctx + (size_t)q * 768 + lane * 8) = hv;
}

// ---------------------------------------------------------------------------
// WMMA GEMM with TDM-staged B panel. Block = 128 rows x 64 cols (8 waves, one
// 16x64 tile each). The packed B panel for a block (4 col-tiles x KB k-blocks)
// is contiguous: 4*KB KB. Wave 0 DMAs it to LDS via tensor_load_to_lds (D# per
// CDNA5 ISA: count=1, type=2, 8-byte elements, 1-D tile), waits on
// TENSORcnt, then all waves read fragments from LDS (ds_load_b128).
// ---------------------------------------------------------------------------
template <int K, int NCOLS, bool RELU>
__global__ __launch_bounds__(256) void gemm_kernel(
    const _Float16* __restrict__ A, const _Float16* __restrict__ Bp,
    const float* __restrict__ bias, _Float16* __restrict__ Out,
    int ldo, int nrows16)
{
    constexpr int KB = K / 32;
    constexpr unsigned PANEL_BYTES = 4u * KB * 1024u;
    extern __shared__ _Float16 lb[];

    int wave = threadIdx.x >> 5;
    int lane = threadIdx.x & 31;
    int m0 = blockIdx.x * 128 + wave * 16;
    int n0 = blockIdx.y * 64;
    int ntile0 = n0 >> 4;

    if (wave == 0) {
        unsigned long long gaddr = (unsigned long long)(Bp + (size_t)ntile0 * KB * 512);
        unsigned lds_off = (unsigned)(size_t)lb;
        unsigned elems = PANEL_BYTES / 8u;  // 8-byte elements, <= 65535
        u32x4 g0;
        g0[0] = 1u;                                    // count=1, user mode
        g0[1] = lds_off;                               // lds_addr (bytes)
        g0[2] = (unsigned)(gaddr & 0xffffffffu);       // global_addr[31:0]
        g0[3] = (unsigned)(gaddr >> 32) | (2u << 30);  // global_addr[56:32] | type=2
        i32x8 g1;
        g1[0] = (int)(3u << 16);                       // data_size = 8B
        g1[1] = (int)((elems & 0xffffu) << 16);        // tensor_dim0[15:0]
        g1[2] = (int)((elems >> 16) & 0xffffu);        // tensor_dim0[31:16]
        g1[3] = (int)((elems & 0xffffu) << 16);        // tile_dim0 = elems
        g1[4] = 0; g1[5] = 0; g1[6] = 0; g1[7] = 0;    // 1-D: no dim1/dim2/strides
        i32x4 z4 = {0, 0, 0, 0};
        i32x8 z8 = {0, 0, 0, 0, 0, 0, 0, 0};
        // clang-23 / therock toolchain: 6-arg form (g0, g1, g2, g3, g4, cpol)
        __builtin_amdgcn_tensor_load_to_lds(g0, g1, z4, z4, z8, 0);
        __builtin_amdgcn_s_wait_tensorcnt(0);
    }
    __syncthreads();
    if (m0 >= nrows16) return;

    v8f acc[4] = {};
    int seg = (lane >> 4) ? 8 : 0;
    int row = m0 + (lane & 15);
    const _Float16* arow  = A + (size_t)row * K + seg;
    const _Float16* blane = lb + (size_t)lane * 16;

    for (int kb = 0; kb < KB; ++kb) {
        const _Float16* ap = arow + kb * 32;
        __builtin_prefetch(ap + 32, 0, 3);  // next K-block of A
        v8h alo = *(const v8h*)(ap);
        v8h ahi = *(const v8h*)(ap + 16);
        v16h av = __builtin_shufflevector(alo, ahi, 0, 1, 2, 3, 4, 5, 6, 7,
                                          8, 9, 10, 11, 12, 13, 14, 15);
#pragma unroll
        for (int s = 0; s < 4; ++s) {
            const v16h bv = *(const v16h*)(blane + (size_t)(s * KB + kb) * 512);
            acc[s] = __builtin_amdgcn_wmma_f32_16x16x32_f16(
                false, av, false, bv, (short)0, acc[s], false, false);
        }
    }

    int rrow  = m0 + ((lane >> 4) ? 8 : 0);
    int cbase = n0 + (lane & 15);
#pragma unroll
    for (int s = 0; s < 4; ++s) {
        int col = cbase + s * 16;
        float bv = bias[col];
#pragma unroll
        for (int r = 0; r < 8; ++r) {
            float v = acc[s][r] + bv;
            if (RELU) v = fmaxf(v, 0.0f);
            Out[(size_t)(rrow + r) * ldo + col] = (_Float16)v;
        }
    }
}

// ---------------------------------------------------------------------------
// Final 256 -> 4 projection in f32 (0.4 GFLOP total; VALU is fine).
// ---------------------------------------------------------------------------
__global__ void final_kernel(const _Float16* __restrict__ H2, const float* __restrict__ W4,
                             const float* __restrict__ b4, float* __restrict__ out, int rows)
{
    int q = blockIdx.x * 256 + threadIdx.x;
    if (q >= rows) return;
    const _Float16* h = H2 + (size_t)q * 256;
    float a0 = b4[0], a1 = b4[1], a2 = b4[2], a3 = b4[3];
    for (int k = 0; k < 256; ++k) {
        float v = (float)h[k];
        a0 += v * W4[4 * k + 0];
        a1 += v * W4[4 * k + 1];
        a2 += v * W4[4 * k + 2];
        a3 += v * W4[4 * k + 3];
    }
    out[4 * q + 0] = a0; out[4 * q + 1] = a1;
    out[4 * q + 2] = a2; out[4 * q + 3] = a3;
}

extern "C" void kernel_launch(void* const* d_in, const int* in_sizes, int n_in,
                              void* d_out, int out_size, void* d_ws, size_t ws_size,
                              hipStream_t stream)
{
    const float* grid  = (const float*)d_in[0];
    const float* sgrad = (const float*)d_in[1];
    const float* qpos  = (const float*)d_in[2];
    const float* quinf = (const float*)d_in[3];
    const float* qsdf  = (const float*)d_in[4];
    const float* qnorm = (const float*)d_in[5];
    const float* qflow = (const float*)d_in[6];
    const float* w1 = (const float*)d_in[7];   // 231 x 256
    const float* b1 = (const float*)d_in[8];
    const float* w2 = (const float*)d_in[9];   // 256 x 512
    const float* b2 = (const float*)d_in[10];
    const float* w3 = (const float*)d_in[11];  // 768 x 256
    const float* b3 = (const float*)d_in[12];
    const float* w4 = (const float*)d_in[13];  // 256 x 4
    const float* b4 = (const float*)d_in[14];
    float* out = (float*)d_out;
    int N = in_sizes[2] / 2;

    // Workspace layout (all offsets 256B aligned)
    char* ws = (char*)d_ws;
    const size_t GT_BYTES = (size_t)HW_ * 256 * 4;  // 128 MB transposed grid
    float*    gt  = (float*)(ws + 0);
    _Float16* w1p = (_Float16*)(ws + GT_BYTES);                      // 131072 B
    _Float16* w2p = (_Float16*)(ws + GT_BYTES + 131072);             // 262144 B
    _Float16* w3p = (_Float16*)(ws + GT_BYTES + 393216);             // 393216 B
    _Float16* ff  = (_Float16*)(ws + GT_BYTES + 786432);
    _Float16* hbf = (_Float16*)(ws + GT_BYTES + 786432 + (size_t)CHUNK * 256 * 2);
    _Float16* ctx = (_Float16*)(ws + GT_BYTES + 786432 + (size_t)CHUNK * 256 * 2 * 2);

    // One-time per launch: transpose grid, pack weights
    transpose_kernel<<<dim3(HW_ / 32, 256 / 32), 256, 0, stream>>>(grid, gt);
    pack_kernel<<<(16 * 8 * 32 + 255) / 256, 256, 0, stream>>>(w1, w1p, 231, 256, 8, 16);
    pack_kernel<<<(32 * 8 * 32 + 255) / 256, 256, 0, stream>>>(w2, w2p, 256, 512, 8, 32);
    pack_kernel<<<(16 * 24 * 32 + 255) / 256, 256, 0, stream>>>(w3, w3p, 768, 256, 24, 16);

    for (int base = 0; base < N; base += CHUNK) {
        int rows = N - base;
        if (rows > CHUNK) rows = CHUNK;
        int rows16 = (rows + 15) & ~15;

        prep_kernel<<<(rows16 + 7) / 8, 256, 0, stream>>>(
            gt, sgrad, qpos, quinf, qsdf, qnorm, qflow, ff, ctx, base, rows, rows16);

        dim3 g1((rows16 + 127) / 128, 256 / 64);
        gemm_kernel<256, 256, true><<<g1, 256, 4 * 8 * 1024, stream>>>(ff, w1p, b1, hbf, 256, rows16);

        dim3 g2((rows16 + 127) / 128, 512 / 64);
        gemm_kernel<256, 512, false><<<g2, 256, 4 * 8 * 1024, stream>>>(hbf, w2p, b2, ctx + 256, 768, rows16);

        dim3 g3((rows16 + 127) / 128, 256 / 64);
        gemm_kernel<768, 256, true><<<g3, 256, 4 * 24 * 1024, stream>>>(ctx, w3p, b3, hbf, 256, rows16);

        final_kernel<<<(rows + 255) / 256, 256, 0, stream>>>(hbf, w4, b4, out + (size_t)base * 4, rows);
    }
}
